// QuantumOncoModel_17626545782826
// MI455X (gfx1250) — matrix operand order repaired and checked by reading, not compiled, not run on hardware
//
#include <hip/hip_runtime.h>
#include <hip/hip_bf16.h>
#include <math.h>

// ---------------------------------------------------------------------------
// CDNA5 (gfx1250) hybrid ResNet-18 + 4-qubit statevector model.
// Convs run as implicit GEMM on v_wmma_f32_16x16x32_f16 (wave32).
// Activations are kept NHWC and GEMM-K is ordered (kh*KW+kw)*Cin + c, so each
// 32-wide K chunk is one filter tap and both A and B fragments load as
// aligned b128 vectors. Wave tile: 32(M) x 64(N) = 8 accumulators; the
// channel-chunk loop is unrolled x2 (chunk count is always even) so the
// scheduler can overlap next-chunk loads with current-chunk WMMAs.
// BN folded into f16 weights + f32 bias; bias/residual/ReLU fused.
// ---------------------------------------------------------------------------

typedef __attribute__((ext_vector_type(16))) _Float16 v16h;
typedef __attribute__((ext_vector_type(8)))  _Float16 v8h;
typedef __attribute__((ext_vector_type(8)))  float    v8f;

#define EPS_BN 1e-5f
#define PI_F   3.14159265358979323846f

// ---------------------------------------------------------------------------
// BN fold with K reorder: wf[(co*KHW + r)*Cin + c] = w[(co*Cin + c)*KHW + r]*sc
// bias[co] = b[co] - m[co]*sc
// ---------------------------------------------------------------------------
__global__ void fold_bn_kernel(const float* __restrict__ w,
                               const float* __restrict__ g,
                               const float* __restrict__ b,
                               const float* __restrict__ m,
                               const float* __restrict__ v,
                               int Cout, int Cin, int KHW,
                               _Float16* __restrict__ wf,
                               float* __restrict__ bias) {
  int Kdim = Cin * KHW;
  int i = blockIdx.x * blockDim.x + threadIdx.x;
  int total = Cout * Kdim;
  if (i < total) {
    int co  = i / Kdim;
    int rem = i - co * Kdim;
    int r   = rem / Cin;
    int c   = rem - r * Cin;
    float sc = g[co] * rsqrtf(v[co] + EPS_BN);
    wf[i] = (_Float16)(w[((size_t)co * Cin + c) * KHW + r] * sc);
  }
  if (i < Cout) {
    float sc = g[i] * rsqrtf(v[i] + EPS_BN);
    bias[i] = b[i] - m[i] * sc;
  }
}

// NCHW f32 -> NHWC f16
__global__ void nchw_to_nhwc_kernel(const float* __restrict__ in,
                                    _Float16* __restrict__ out,
                                    int NB, int C, int H, int W) {
  int idx = blockIdx.x * blockDim.x + threadIdx.x;
  int total = NB * C * H * W;
  if (idx >= total) return;
  int c = idx % C;
  int t = idx / C;
  int w = t % W;  t /= W;
  int h = t % H;
  int n = t / H;
  out[idx] = (_Float16)in[(((size_t)n * C + c) * H + h) * W + w];
}

// ---------------------------------------------------------------------------
// Implicit-GEMM conv with WMMA, 32x64 tile per wave (8 accumulators), NHWC.
//
// Fragment layouts per CDNA5 ISA 7.12.2 (16-bit, wave32):
//  A (16x32, MxK): lane = hi*16 + laneM; element e -> K = k0 + hi*16 + e (contiguous, not the ISA table order)
//  B (32x16, KxN): lane holds N = lane&15; element e -> K = k0 + hi*16 + e (same order as A; not the ISA table order)
//  C/D (16x16 f32): VGPR r -> M = r + hi*8, N = lane&15
//
// K ordering: K = (kh*KW + kw)*Cin + c  (weights folded in this layout).
// NHWC: A fragment = two contiguous 8xf16 runs per lane -> 2 aligned b128.
// Guarantees: M = 64*P*Q multiple of 32; Cout multiple of 64; for !GK paths
// Cin is a multiple of 32 (chunk count even) so each 32-chunk has one (kh,kw).
// ---------------------------------------------------------------------------
template <int KH, int KW, int CIN_CT, bool GK>
__global__ __launch_bounds__(256) void conv_wmma_kernel(
    const _Float16* __restrict__ in,   // (N, H, W, Cin)  NHWC
    const _Float16* __restrict__ wt,   // (Cout, KHW*Cin) reordered
    const float*    __restrict__ bias, // (Cout)
    const _Float16* __restrict__ resid,// (N, P, Q, Cout) or aliased to out
    _Float16*       __restrict__ out,  // (N, P, Q, Cout) NHWC
    int N, int Cin, int H, int W, int Cout, int P, int Q,
    int stride, int pad, int relu, int has_res) {
  constexpr int KHW = KH * KW;
  const int M    = N * P * Q;          // multiple of 32
  const int Kdim = Cin * KHW;
  const int PQ   = P * Q;
  const int tilesM = M >> 5;           // 32-row M blocking
  const int tilesN = Cout >> 6;        // 64-col N blocking

  int wave = blockIdx.x * (blockDim.x >> 5) + (threadIdx.x >> 5);
  if (wave >= tilesM * tilesN) return;  // wave-uniform: EXEC stays full
  int tm = wave % tilesM;
  int tn = wave / tilesM;

  int lane  = threadIdx.x & 31;
  int laneM = lane & 15;
  int hi    = lane >> 4;

  // A-side rows for the two 16-row subtiles
  int ih0[2], iw0[2];
  const _Float16* inN[2];
#pragma unroll
  for (int mb = 0; mb < 2; ++mb) {
    int row = tm * 32 + mb * 16 + laneM;
    int n0  = row / PQ;
    int pq0 = row - n0 * PQ;
    int p0  = pq0 / Q;
    int q0  = pq0 - p0 * Q;
    ih0[mb] = p0 * stride - pad;
    iw0[mb] = q0 * stride - pad;
    inN[mb] = in + (size_t)n0 * H * W * Cin;
  }

  // B-side: 4 column subtiles, 16 apart
  int colBase = tn * 64 + laneM;
  const _Float16* wrow[4];
#pragma unroll
  for (int s = 0; s < 4; ++s) wrow[s] = wt + (size_t)(colBase + s * 16) * Kdim;

  v8f acc[4][2];
#pragma unroll
  for (int s = 0; s < 4; ++s) {
    acc[s][0] = v8f{};
    acc[s][1] = v8f{};
  }

  if constexpr (!GK) {
    // ---- Fast path: each 32-chunk lies in one filter tap; NHWC vector A ----
#pragma unroll
    for (int r = 0; r < KHW; ++r) {
      const int kh = r / KW;
      const int kw = r - kh * KW;
      bool ok[2];
      const _Float16* abase[2];
#pragma unroll
      for (int mb = 0; mb < 2; ++mb) {
        int ih = ih0[mb] + kh;
        int iw = iw0[mb] + kw;
        ok[mb] = (KH == 1) ? true
                           : ((unsigned)ih < (unsigned)H && (unsigned)iw < (unsigned)W);
        abase[mb] = inN[mb] + (long)(ih * W + iw) * Cin + (hi << 4);
      }
      // chunk count Cin/32 is even -> unroll x2 exact; lets the scheduler
      // overlap next-chunk loads with current-chunk WMMAs.
#pragma unroll 2
      for (int c0 = 0; c0 < Cin; c0 += 32) {
        // A fragments: two aligned 16B vector loads per lane
        v16h a[2];
#pragma unroll
        for (int mb = 0; mb < 2; ++mb) {
          v8h lo = {}, hh = {};
          if (ok[mb]) {
            const _Float16* ap = abase[mb] + c0;
            lo = *(const v8h*)ap;          // channels c0+hi*16 + 0..7
            hh = *(const v8h*)(ap + 8);    // channels c0+hi*16 + 8..15
          }
#pragma unroll
          for (int e = 0; e < 8; ++e) {
            a[mb][e]     = lo[e];
            a[mb][e + 8] = hh[e];
          }
        }
        // B fragments: aligned 32B vector loads + prefetch of next chunk
        int kb = r * Cin + c0 + (hi << 4);
        v16h bfr[4];
#pragma unroll
        for (int s = 0; s < 4; ++s) {
          bfr[s] = *(const v16h*)(wrow[s] + kb);
          __builtin_prefetch(wrow[s] + kb + 32, 0, 1);
        }
#pragma unroll
        for (int s = 0; s < 4; ++s)
#pragma unroll
          for (int mb = 0; mb < 2; ++mb)
            acc[s][mb] = __builtin_amdgcn_wmma_f32_16x16x32_f16(
                false, a[mb], false, bfr[s], (short)0, acc[s][mb], false, false);
      }
    }
  } else {
    // ---- Stem path (Cin = CIN_CT compile-time, Kdim not 32-aligned) ----
    for (int k0 = 0; k0 < Kdim; k0 += 32) {
      v16h a[2];
#pragma unroll
      for (int mb = 0; mb < 2; ++mb) {
#pragma unroll
        for (int e = 0; e < 16; ++e) {
          int k = k0 + (hi << 4) + e;
          _Float16 av = (_Float16)0.0f;
          if (k < Kdim) {
            int r  = k / CIN_CT;       // compile-time divisor
            int c  = k - r * CIN_CT;
            int kh = r / KW;
            int kw = r - kh * KW;
            int ih = ih0[mb] + kh;
            int iw = iw0[mb] + kw;
            if ((unsigned)ih < (unsigned)H && (unsigned)iw < (unsigned)W)
              av = inN[mb][(long)(ih * W + iw) * CIN_CT + c];
          }
          a[mb][e] = av;
        }
      }
      v16h bfr[4];
#pragma unroll
      for (int s = 0; s < 4; ++s) {
#pragma unroll
        for (int e = 0; e < 16; ++e) {
          int k = k0 + (hi << 4) + e;
          bfr[s][e] = (k < Kdim) ? wrow[s][k] : (_Float16)0.0f;
        }
      }
#pragma unroll
      for (int s = 0; s < 4; ++s)
#pragma unroll
        for (int mb = 0; mb < 2; ++mb)
          acc[s][mb] = __builtin_amdgcn_wmma_f32_16x16x32_f16(
              false, a[mb], false, bfr[s], (short)0, acc[s][mb], false, false);
    }
  }

  // ---- Epilogue: bias + residual + ReLU, store f16 (NHWC, lane=channel) ----
  float bv[4];
#pragma unroll
  for (int s = 0; s < 4; ++s) bv[s] = bias[colBase + s * 16];

#pragma unroll
  for (int mb = 0; mb < 2; ++mb) {
#pragma unroll
    for (int r = 0; r < 8; ++r) {
      int mrow = tm * 32 + mb * 16 + (hi << 3) + r;
      size_t pixBase = (size_t)mrow * Cout;  // NHWC: (n*PQ + pq)*Cout
#pragma unroll
      for (int s = 0; s < 4; ++s) {
        size_t oidx = pixBase + (size_t)(colBase + s * 16);
        float vv = acc[s][mb][r] + bv[s];
        if (has_res) vv += (float)resid[oidx];
        if (relu) vv = fmaxf(vv, 0.0f);
        out[oidx] = (_Float16)vv;
      }
    }
  }
}

// ---------------------------------------------------------------------------
// 3x3 stride-2 pad-1 max pool, NHWC (pad value = -inf per reduce_window init)
// ---------------------------------------------------------------------------
__global__ void maxpool_kernel(const _Float16* __restrict__ in,
                               _Float16* __restrict__ out,
                               int NB, int C, int H, int W, int P, int Q) {
  int idx = blockIdx.x * blockDim.x + threadIdx.x;
  int total = NB * C * P * Q;
  if (idx >= total) return;
  int c = idx % C;
  int t = idx / C;
  int q = t % Q;  t /= Q;
  int p = t % P;
  int n = t / P;
  float mx = -3.4e38f;
  for (int kh = 0; kh < 3; ++kh)
    for (int kw = 0; kw < 3; ++kw) {
      int ih = p * 2 - 1 + kh;
      int iw = q * 2 - 1 + kw;
      if ((unsigned)ih < (unsigned)H && (unsigned)iw < (unsigned)W) {
        float v = (float)in[(((size_t)n * H + ih) * W + iw) * C + c];
        mx = fmaxf(mx, v);
      }
    }
  out[idx] = (_Float16)mx;
}

// ---------------------------------------------------------------------------
// Global average pool (7x7x512 NHWC) + FC(512->4) + sigmoid*pi -> angles
// ---------------------------------------------------------------------------
__global__ __launch_bounds__(512) void pool_fc_kernel(
    const _Float16* __restrict__ in,  // (64, 7, 7, 512) NHWC
    const float* __restrict__ fcw,    // (4, 512)
    const float* __restrict__ fcb,    // (4)
    float* __restrict__ angles) {     // (64, 4)
  __shared__ float feat[512];
  int n = blockIdx.x;
  int c = threadIdx.x;
  float sum = 0.0f;
  const _Float16* src = in + (size_t)n * 49 * 512 + c;
  for (int j = 0; j < 49; ++j) sum += (float)src[(size_t)j * 512];
  feat[c] = sum * (1.0f / 49.0f);
  __syncthreads();
  if (c < 4) {
    float acc = fcb[c];
    for (int k = 0; k < 512; ++k) acc += fcw[c * 512 + k] * feat[k];
    float sig = 1.0f / (1.0f + expf(-acc));
    angles[n * 4 + c] = sig * PI_F;
  }
}

// ---------------------------------------------------------------------------
// 4-qubit statevector: AngleEmbedding(RX) + 3x StronglyEntanglingLayers,
// output (<Z0>+1)/2. One thread per batch element; 16 complex amps in regs.
// Qubit 0 = MSB of the 4-bit index (einsum a,b,c,d -> abcd).
// ---------------------------------------------------------------------------
__global__ void quantum_kernel(const float* __restrict__ angles, // (64,4)
                               const float* __restrict__ qw,     // (3,4,3)
                               float* __restrict__ out) {        // (64)
  int i = blockIdx.x * blockDim.x + threadIdx.x;
  if (i >= 64) return;

  float cq[4], sq[4];
#pragma unroll
  for (int q = 0; q < 4; ++q) {
    float a = angles[i * 4 + q] * 0.5f;
    cq[q] = cosf(a);
    sq[q] = sinf(a);
  }

  float re[16], im[16];
#pragma unroll
  for (int idx = 0; idx < 16; ++idx) {
    float ar = 1.0f, ai = 0.0f;
#pragma unroll
    for (int q = 0; q < 4; ++q) {
      int bit = (idx >> (3 - q)) & 1;
      float fr = bit ? 0.0f : cq[q];
      float fi = bit ? -sq[q] : 0.0f;
      float nr = ar * fr - ai * fi;
      float ni = ar * fi + ai * fr;
      ar = nr; ai = ni;
    }
    re[idx] = ar; im[idx] = ai;
  }

  for (int l = 0; l < 3; ++l) {
    // Rot on each wire: RZ(omega) RY(theta) RZ(phi)
    for (int w = 0; w < 4; ++w) {
      const float* ww = qw + (l * 4 + w) * 3;
      float phi = ww[0], th = ww[1], om = ww[2];
      float ct = cosf(th * 0.5f), st = sinf(th * 0.5f);
      float h1 = 0.5f * (phi + om), h2 = 0.5f * (phi - om);
      float c1 = cosf(h1), s1 = sinf(h1);
      float c2 = cosf(h2), s2 = sinf(h2);
      float u00r =  c1 * ct, u00i = -s1 * ct;
      float u01r = -c2 * st, u01i = -s2 * st;
      float u10r =  c2 * st, u10i = -s2 * st;
      float u11r =  c1 * ct, u11i =  s1 * ct;
      int msk = 1 << (3 - w);
#pragma unroll
      for (int idx = 0; idx < 16; ++idx) {
        if (idx & msk) continue;
        int j = idx | msk;
        float a0r = re[idx], a0i = im[idx];
        float a1r = re[j],   a1i = im[j];
        re[idx] = u00r * a0r - u00i * a0i + u01r * a1r - u01i * a1i;
        im[idx] = u00r * a0i + u00i * a0r + u01r * a1i + u01i * a1r;
        re[j]   = u10r * a0r - u10i * a0i + u11r * a1r - u11i * a1i;
        im[j]   = u10r * a0i + u10i * a0r + u11r * a1i + u11i * a1r;
      }
    }
    // CNOT ring, range r = l % 3 + 1
    int rrng = l % 3 + 1;
    for (int w = 0; w < 4; ++w) {
      int cm = 1 << (3 - w);
      int tg = 1 << (3 - ((w + rrng) & 3));
#pragma unroll
      for (int idx = 0; idx < 16; ++idx) {
        if ((idx & cm) && !(idx & tg)) {
          int j = idx | tg;
          float tr = re[idx]; re[idx] = re[j]; re[j] = tr;
          float ti = im[idx]; im[idx] = im[j]; im[j] = ti;
        }
      }
    }
  }

  float z = 0.0f;
#pragma unroll
  for (int idx = 0; idx < 16; ++idx) {
    float p = re[idx] * re[idx] + im[idx] * im[idx];
    z += (idx & 8) ? -p : p;
  }
  out[i] = (z + 1.0f) * 0.5f;
}

// ---------------------------------------------------------------------------
// Host orchestration
// ---------------------------------------------------------------------------
struct BNp { const float *g, *b, *m, *v; };
struct ConvDesc { const float* w; BNp bn; int cout, cin, k; };

static inline int cdiv(long a, long b) { return (int)((a + b - 1) / b); }

static void launch_conv(hipStream_t s,
                        const _Float16* in, int N, int Cin, int H, int W,
                        const _Float16* w, const float* bias,
                        int Cout, int K, int stride, int pad,
                        const _Float16* resid, int relu,
                        _Float16* out, int* Pout, int* Qout) {
  int P = (H + 2 * pad - K) / stride + 1;
  int Q = (W + 2 * pad - K) / stride + 1;
  *Pout = P; *Qout = Q;
  long tiles = ((long)N * P * Q / 32) * (Cout / 64);
  int blocks = cdiv(tiles, 8);  // 8 waves / 256-thread block
  const _Float16* res = resid ? resid : out;
  int hr = resid ? 1 : 0;
  if (K == 7)
    conv_wmma_kernel<7, 7, 3, true><<<blocks, 256, 0, s>>>(
        in, w, bias, res, out, N, Cin, H, W, Cout, P, Q, stride, pad, relu, hr);
  else if (K == 3)
    conv_wmma_kernel<3, 3, 0, false><<<blocks, 256, 0, s>>>(
        in, w, bias, res, out, N, Cin, H, W, Cout, P, Q, stride, pad, relu, hr);
  else
    conv_wmma_kernel<1, 1, 0, false><<<blocks, 256, 0, s>>>(
        in, w, bias, res, out, N, Cin, H, W, Cout, P, Q, stride, pad, relu, hr);
}

extern "C" void kernel_launch(void* const* d_in, const int* in_sizes, int n_in,
                              void* d_out, int out_size, void* d_ws, size_t ws_size,
                              hipStream_t stream) {
  (void)in_sizes; (void)n_in; (void)out_size; (void)ws_size;
  const int NB = 64;

  // ---- Parse d_in in setup_inputs() insertion order ----
  int ii = 0;
  const float* x = (const float*)d_in[ii++];

  auto getBN = [&](void) {
    BNp p;
    p.g = (const float*)d_in[ii++];
    p.b = (const float*)d_in[ii++];
    p.m = (const float*)d_in[ii++];
    p.v = (const float*)d_in[ii++];
    return p;
  };

  ConvDesc convs[20];
  int nc = 0;

  // stem
  convs[nc].w = (const float*)d_in[ii++];
  convs[nc].bn = getBN();
  convs[nc].cout = 64; convs[nc].cin = 3; convs[nc].k = 7;
  int stemIdx = nc++;

  struct BlockIdx { int c1, c2, cd; };
  BlockIdx blocks[8];
  int bi = 0;
  const int cins[4]    = {64, 64, 128, 256};
  const int couts[4]   = {64, 128, 256, 512};
  const int strides[4] = {1, 2, 2, 2};
  for (int st = 0; st < 4; ++st) {
    for (int b = 0; b < 2; ++b) {
      int cin = (b == 0) ? cins[st] : couts[st];
      bool down = (b == 0) && (strides[st] != 1 || cin != couts[st]);
      BlockIdx bl;
      bl.c1 = nc;
      convs[nc].w = (const float*)d_in[ii++]; convs[nc].bn = getBN();
      convs[nc].cout = couts[st]; convs[nc].cin = cin; convs[nc].k = 3; nc++;
      bl.c2 = nc;
      convs[nc].w = (const float*)d_in[ii++]; convs[nc].bn = getBN();
      convs[nc].cout = couts[st]; convs[nc].cin = couts[st]; convs[nc].k = 3; nc++;
      if (down) {
        bl.cd = nc;
        convs[nc].w = (const float*)d_in[ii++]; convs[nc].bn = getBN();
        convs[nc].cout = couts[st]; convs[nc].cin = cin; convs[nc].k = 1; nc++;
      } else bl.cd = -1;
      blocks[bi++] = bl;
    }
  }
  const float* fcw = (const float*)d_in[ii++];
  const float* fcb = (const float*)d_in[ii++];
  const float* qw  = (const float*)d_in[ii++];

  // ---- Carve workspace ----
  char* wsp = (char*)d_ws;
  auto carve = [&](size_t bytes) {
    char* p = wsp;
    wsp += (bytes + 255) & ~(size_t)255;
    return p;
  };
  const size_t X_ELEMS = (size_t)NB * 3 * 224 * 224;         // 9,633,792
  const size_t BIG     = (size_t)NB * 64 * 112 * 112;        // 51,380,224
  const size_t MED     = (size_t)NB * 64 * 56 * 56;          // 12,845,056
  _Float16* x16  = (_Float16*)carve(X_ELEMS * 2);
  _Float16* bufA = (_Float16*)carve(BIG * 2);
  _Float16* bufB = (_Float16*)carve(BIG * 2);
  _Float16* bufC = (_Float16*)carve(MED * 2);
  _Float16* wf[20];
  float*    bf[20];
  for (int i = 0; i < nc; ++i) {
    size_t kd = (size_t)convs[i].cin * convs[i].k * convs[i].k;
    wf[i] = (_Float16*)carve((size_t)convs[i].cout * kd * 2);
    bf[i] = (float*)carve((size_t)convs[i].cout * 4);
  }
  float* angles = (float*)carve((size_t)NB * 4 * 4);

  // ---- Preprocess: fold BN (with K reorder), x -> NHWC f16 ----
  for (int i = 0; i < nc; ++i) {
    int khw = convs[i].k * convs[i].k;
    int tot = convs[i].cout * convs[i].cin * khw;
    fold_bn_kernel<<<cdiv(tot, 256), 256, 0, stream>>>(
        convs[i].w, convs[i].bn.g, convs[i].bn.b, convs[i].bn.m, convs[i].bn.v,
        convs[i].cout, convs[i].cin, khw, wf[i], bf[i]);
  }
  nchw_to_nhwc_kernel<<<cdiv((long)X_ELEMS, 256), 256, 0, stream>>>(
      x, x16, NB, 3, 224, 224);

  // ---- Stem: conv7x7 s2 p3 + BN + ReLU, then maxpool ----
  int P, Q;
  launch_conv(stream, x16, NB, 3, 224, 224, wf[stemIdx], bf[stemIdx],
              64, 7, 2, 3, nullptr, 1, bufA, &P, &Q);           // (64,112,112,64)
  {
    int total = NB * 64 * 56 * 56;
    maxpool_kernel<<<cdiv(total, 256), 256, 0, stream>>>(bufA, bufB, NB, 64, 112, 112, 56, 56);
  }

  // ---- Residual stages ----
  _Float16* cur = bufB;
  _Float16* freeBuf[2] = {bufA, bufC};
  int curC = 64, curH = 56, curW = 56;

  bi = 0;
  for (int st = 0; st < 4; ++st) {
    for (int b = 0; b < 2; ++b) {
      BlockIdx bl = blocks[bi++];
      int s = (b == 0) ? strides[st] : 1;
      const ConvDesc& c1 = convs[bl.c1];
      const ConvDesc& c2 = convs[bl.c2];

      _Float16* out1 = freeBuf[0];
      int P1, Q1;
      launch_conv(stream, cur, NB, curC, curH, curW, wf[bl.c1], bf[bl.c1],
                  c1.cout, 3, s, 1, nullptr, 1, out1, &P1, &Q1);

      if (bl.cd >= 0) {
        _Float16* sc = freeBuf[1];
        int Pd, Qd;
        launch_conv(stream, cur, NB, curC, curH, curW, wf[bl.cd], bf[bl.cd],
                    convs[bl.cd].cout, 1, s, 0, nullptr, 0, sc, &Pd, &Qd);
        int P2, Q2;
        launch_conv(stream, out1, NB, c1.cout, P1, Q1, wf[bl.c2], bf[bl.c2],
                    c2.cout, 3, 1, 1, sc, 1, sc, &P2, &Q2);
        // rotate buffers: sc becomes current
        _Float16* oldCur = cur;
        cur = sc;
        freeBuf[0] = oldCur;
        freeBuf[1] = out1;
        curC = c2.cout; curH = P2; curW = Q2;
      } else {
        int P2, Q2;
        launch_conv(stream, out1, NB, c1.cout, P1, Q1, wf[bl.c2], bf[bl.c2],
                    c2.cout, 3, 1, 1, cur, 1, cur, &P2, &Q2);
        curC = c2.cout; curH = P2; curW = Q2;
        // cur unchanged; freeBuf unchanged
      }
    }
  }

  // ---- Head: avgpool + FC + sigmoid*pi, then quantum circuit ----
  pool_fc_kernel<<<NB, 512, 0, stream>>>(cur, fcw, fcb, angles);
  quantum_kernel<<<1, 64, 0, stream>>>(angles, qw, (float*)d_out);
}
